// EncoderDecoderTransformer_66769561584317
// MI455X (gfx1250) — compile-verified, matrix operand order checked by reading, not run on hardware
//
#include <hip/hip_runtime.h>

// ---------------------------------------------------------------------------
// Problem constants (from reference)
// ---------------------------------------------------------------------------
#define D_    512
#define F_    2048
#define H_    8
#define NE_   4
#define ND_   4
#define KNN_  16
#define B_    4
#define LE_   1024
#define LD_   1024
#define MTOK  (B_ * LE_)          // 4096 token rows (B*L) for both enc and dec

// ---------------------------------------------------------------------------
// CDNA5 WMMA helpers (gfx1250, wave32)
// ---------------------------------------------------------------------------
typedef __attribute__((ext_vector_type(16))) _Float16 v16h;
typedef __attribute__((ext_vector_type(8)))  _Float16 v8h;
typedef __attribute__((ext_vector_type(8)))  float    v8f;

__device__ __forceinline__ v8f wmma_f16(v16h a, v16h b, v8f c) {
  // D = A(16x32 f16) x B(32x16 f16) + C(16x16 f32)
  return __builtin_amdgcn_wmma_f32_16x16x32_f16(false, a, false, b,
                                                (short)0, c, false, false);
}

__device__ __forceinline__ float gelu_exact(float x) {
  return 0.5f * x * (1.0f + erff(x * 0.70710678118654752f));
}

// Async global->LDS staging (gfx1250 ASYNCcnt path), feature-gated so the
// build falls back to plain vector copies if the builtins are absent.
// Builtin signature (from hipcc diagnostic): pointers to int4 vectors in
// AS1 (global) / AS3 (LDS).
#if defined(__AMDGCN__) && \
    __has_builtin(__builtin_amdgcn_global_load_async_to_lds_b128) && \
    __has_builtin(__builtin_amdgcn_s_wait_asynccnt)
#define USE_ASYNC_LDS 1
typedef int v4i_gcc __attribute__((vector_size(4 * sizeof(int))));
#else
#define USE_ASYNC_LDS 0
#endif

// Copy 8 f16 (16B) from global to LDS: async DMA if available.
__device__ __forceinline__ void cp8h(_Float16* lds, const _Float16* g) {
#if USE_ASYNC_LDS
  __builtin_amdgcn_global_load_async_to_lds_b128(
      (__attribute__((address_space(1))) v4i_gcc*)(g),
      (__attribute__((address_space(3))) v4i_gcc*)(lds), 0, 0);
#else
  *(v8h*)lds = *(const v8h*)g;
#endif
}

// ===========================================================================
// f32 -> f16 conversion (weights, once per launch)
// ===========================================================================
__global__ __launch_bounds__(256) void cvt_kernel(const float* __restrict__ src,
                                                  _Float16* __restrict__ dst,
                                                  int n) {
  int i = (blockIdx.x * 256 + threadIdx.x) * 4;
  if (i >= n) return;
  float4 v = *(const float4*)(src + i);
  dst[i + 0] = (_Float16)v.x;
  dst[i + 1] = (_Float16)v.y;
  dst[i + 2] = (_Float16)v.z;
  dst[i + 3] = (_Float16)v.w;
}

// ===========================================================================
// Generic GEMM:  C[M,N] = act(A[M,K] * W[N,K]^T + bias[N]) (+ residual)
// A, W are f16; accumulate f32; output f32 (Cf) or f16 (Ch).
// Block: 256 threads (8 waves).  Block tile 64(M) x 256(N), K-step 32.
// Wave (wm,wn) owns 32x64 = 2x4 WMMA accumulators.
// Double-buffered LDS staged by async DMA: each thread issues 5 b128 copies
// per tile, so s_wait_asynccnt(5) after issuing tile t+1 fences tile t ->
// DMA of the next tile overlaps the 8 WMMAs of the current one.
// Requires: M%64==0, N%256==0, K%32==0 (true for all uses here).
// ===========================================================================
#define GBM 64
#define GBN 256
#define PAD_H 40   // f16 row stride: 80B = 20 banks -> 16 rows hit 16 banks

__global__ __launch_bounds__(256) void gemm_kernel(
    const _Float16* __restrict__ A, int lda,
    const _Float16* __restrict__ W, int Kdim,     // N x Kdim row-major (f16)
    const float* __restrict__ bias,               // length N (f32)
    const float* __restrict__ res,                // nullable, stride ldc (f32)
    float* __restrict__ Cf, _Float16* __restrict__ Ch,  // one non-null
    int ldc, int act)                             // act: 0=none, 1=gelu
{
  __shared__ _Float16 As[2][GBM][PAD_H];
  __shared__ _Float16 Bs[2][GBN][PAD_H];

  const int tid = threadIdx.x;
  const int w   = tid >> 5;
  const int l   = tid & 31;
  const int wm  = w & 1;          // 0..1  -> rows wm*32
  const int wn  = w >> 1;         // 0..3  -> cols wn*64
  const int mbase = blockIdx.y * GBM;
  const int nbase = blockIdx.x * GBN;

  v8f acc[2][4];
#pragma unroll
  for (int mt = 0; mt < 2; ++mt)
#pragma unroll
    for (int nt = 0; nt < 4; ++nt) acc[mt][nt] = (v8f){};

  const int lr = l & 15;
  const int ho = (l >= 16) ? 8 : 0;
  const int ra = tid >> 2;          // A chunk row (1 chunk/thread)
  const int ca = (tid & 3) * 8;     // A chunk col (8 halves)

  const int nk = Kdim / 32;

  // Stage tile kb into buffer buf (5 b128 copies per thread)
  auto stage = [&](int kb, int buf) {
    cp8h(&As[buf][ra][ca], A + (size_t)(mbase + ra) * lda + kb + ca);
#pragma unroll
    for (int i = 0; i < 4; ++i) {
      int ch = tid + 256 * i;
      int rr = ch >> 2, cc = (ch & 3) * 8;
      cp8h(&Bs[buf][rr][cc], W + (size_t)(nbase + rr) * Kdim + kb + cc);
    }
  };

  stage(0, 0);
  for (int t = 0; t < nk; ++t) {
    const int buf = t & 1;
    if (t + 1 < nk) stage((t + 1) * 32, buf ^ 1);
#if USE_ASYNC_LDS
    if (t + 1 < nk) __builtin_amdgcn_s_wait_asynccnt(5);  // tile t resident
    else            __builtin_amdgcn_s_wait_asynccnt(0);
#endif
    __syncthreads();

    v16h af[2], bf[4];
#pragma unroll
    for (int mt = 0; mt < 2; ++mt) {
      int row = wm * 32 + mt * 16 + lr;
#pragma unroll
      for (int j = 0; j < 8; ++j) {
        af[mt][j]     = As[buf][row][ho + j];
        af[mt][8 + j] = As[buf][row][16 + ho + j];
      }
    }
#pragma unroll
    for (int nt = 0; nt < 4; ++nt) {
      int row = wn * 64 + nt * 16 + lr;
#pragma unroll
      for (int j = 0; j < 8; ++j) {
        bf[nt][j]     = Bs[buf][row][ho + j];
        bf[nt][8 + j] = Bs[buf][row][16 + ho + j];
      }
    }
#pragma unroll
    for (int mt = 0; mt < 2; ++mt)
#pragma unroll
      for (int nt = 0; nt < 4; ++nt)
        acc[mt][nt] = wmma_f16(af[mt], bf[nt], acc[mt][nt]);
    __syncthreads();   // protect buf before it is re-staged at t+1
  }

  // Epilogue: bias (+gelu) (+residual) -> f32 or f16
#pragma unroll
  for (int mt = 0; mt < 2; ++mt) {
#pragma unroll
    for (int nt = 0; nt < 4; ++nt) {
      int n  = nbase + wn * 64 + nt * 16 + lr;
      float bv = bias[n];
#pragma unroll
      for (int r = 0; r < 8; ++r) {
        int m = mbase + wm * 32 + mt * 16 + r + ho;
        float v = acc[mt][nt][r] + bv;
        if (act == 1) v = gelu_exact(v);
        if (res) v += res[(size_t)m * ldc + n];
        if (Cf) Cf[(size_t)m * ldc + n] = v;
        else    Ch[(size_t)m * ldc + n] = (_Float16)v;
      }
    }
  }
}

// ===========================================================================
// LayerNorm over rows of width 512.  One wave per row, 8 rows per block.
// Optional f32 and/or f16 outputs.
// ===========================================================================
__global__ __launch_bounds__(256) void ln_kernel(
    const float* __restrict__ x, const float* __restrict__ w,
    const float* __restrict__ b, float* __restrict__ outf,
    _Float16* __restrict__ outh, int M)
{
  const int wid = threadIdx.x >> 5;
  const int l   = threadIdx.x & 31;
  const int row = blockIdx.x * 8 + wid;
  if (row >= M) return;
  const float* xr = x + (size_t)row * D_;

  float v[16];
  float s = 0.f;
#pragma unroll
  for (int j = 0; j < 16; ++j) { v[j] = xr[l + 32 * j]; s += v[j]; }
#pragma unroll
  for (int m = 16; m >= 1; m >>= 1) s += __shfl_xor(s, m, 32);
  float mean = s * (1.0f / (float)D_);

  float var = 0.f;
#pragma unroll
  for (int j = 0; j < 16; ++j) { float d = v[j] - mean; var += d * d; }
#pragma unroll
  for (int m = 16; m >= 1; m >>= 1) var += __shfl_xor(var, m, 32);
  var *= (1.0f / (float)D_);
  float inv = rsqrtf(var + 1e-5f);

#pragma unroll
  for (int j = 0; j < 16; ++j) {
    int c = l + 32 * j;
    float o = (v[j] - mean) * inv * w[c] + b[c];
    if (outf) outf[(size_t)row * D_ + c] = o;
    if (outh) outh[(size_t)row * D_ + c] = (_Float16)o;
  }
}

// ===========================================================================
// kNN allow-mask: per (b,q) row, bits for {16 nearest by distance} | self |
// col 0; row 0 allows everything.  bit set == attention allowed.
// ===========================================================================
__global__ void knn_kernel(const float* __restrict__ xyz,
                           unsigned* __restrict__ bits)
{
  int i = blockIdx.x * blockDim.x + threadIdx.x;
  if (i >= B_ * LE_) return;
  int b = i >> 10;
  int q = i & (LE_ - 1);
  const float* base = xyz + (size_t)b * LE_ * 3;
  float qx = base[q * 3 + 0], qy = base[q * 3 + 1], qz = base[q * 3 + 2];

  float bd[KNN_];
  int   bj[KNN_];
#pragma unroll
  for (int t = 0; t < KNN_; ++t) { bd[t] = 3.0e38f; bj[t] = 0; }

  for (int j = 0; j < LE_; ++j) {
    if (j == q) continue;
    float dx = base[j * 3 + 0] - qx;
    float dy = base[j * 3 + 1] - qy;
    float dz = base[j * 3 + 2] - qz;
    float d2 = dx * dx + dy * dy + dz * dz;
    if (d2 < bd[KNN_ - 1]) {
      int p = KNN_ - 1;
      while (p > 0 && bd[p - 1] > d2) { bd[p] = bd[p - 1]; bj[p] = bj[p - 1]; --p; }
      bd[p] = d2; bj[p] = j;
    }
  }

  unsigned wbuf[LE_ / 32];
  if (q == 0) {
#pragma unroll
    for (int t = 0; t < LE_ / 32; ++t) wbuf[t] = 0xFFFFFFFFu;
  } else {
#pragma unroll
    for (int t = 0; t < LE_ / 32; ++t) wbuf[t] = 0u;
    for (int t = 0; t < KNN_; ++t) wbuf[bj[t] >> 5] |= 1u << (bj[t] & 31);
    wbuf[q >> 5] |= 1u << (q & 31);   // self
    wbuf[0]      |= 1u;               // BOS column
  }
  unsigned* orow = bits + (size_t)i * (LE_ / 32);
  for (int t = 0; t < LE_ / 32; ++t) orow[t] = wbuf[t];
}

// ===========================================================================
// Fused flash attention (dh=64, H=8), all operands f16, accum f32.
// One block = (b, h, 32-query tile), 256 threads = 8 waves.
// Key chunks of 256, online softmax.  maskType: 0=none,1=causal,2=bitmask.
// ===========================================================================
#define QT 32
#define CH 256
#define QS_P 72    // 144B rows
#define SS_P 260   // 1040B rows
#define PS_P 264   // 528B rows

__global__ __launch_bounds__(256) void attn_kernel(
    const _Float16* __restrict__ qp, int qs,
    const _Float16* __restrict__ kp, int ks,
    const _Float16* __restrict__ vp, int vs,
    _Float16* __restrict__ op, int os,
    int Lq, int Lk, int maskType,
    const unsigned* __restrict__ mbits, float scale)
{
  __shared__ _Float16 Qs[QT][QS_P];
  __shared__ float    Ss[QT][SS_P];
  __shared__ _Float16 Ps[QT][PS_P];
  __shared__ float    red[QT][8];
  __shared__ float    rowM[QT], rowL[QT], rowSc[QT];

  const int tid = threadIdx.x;
  const int w   = tid >> 5;
  const int l   = tid & 31;
  const int b   = blockIdx.z;
  const int h   = blockIdx.y;
  const int qbase = blockIdx.x * QT;
  const int hoff  = h * 64;

  // Stage Q tile (32x64 f16) to LDS: 1 b128 chunk per thread (async DMA).
  {
    int r = tid >> 3, c8 = (tid & 7) * 8;
    cp8h(&Qs[r][c8], qp + (size_t)(b * Lq + qbase + r) * qs + hoff + c8);
  }
  if (tid < QT) { rowM[tid] = -1e30f; rowL[tid] = 0.f; }
#if USE_ASYNC_LDS
  __builtin_amdgcn_s_wait_asynccnt(0);
#endif
  __syncthreads();

  const int mt = w & 1;        // this wave's O tile: rows mt*16..
  const int nt = w >> 1;       // cols nt*16..  (nt in 0..3)
  const int lr = l & 15;
  const int ho = (l >= 16) ? 8 : 0;
  v8f co = (v8f){};

  const int nch   = Lk / CH;
  const int row   = tid >> 3;  // softmax row (0..31)
  const int sub   = tid & 7;   // 8 threads per row
  const int mwrds = Lk / 32;

  for (int c = 0; c < nch; ++c) {
    const int k0 = c * CH;
    if (maskType == 1 && k0 > qbase + QT - 1) break;   // causal: rest masked

    // ---- S chunk: 32 (smt,snt) 16x16 tiles, 4 per wave -------------------
#pragma unroll
    for (int p = 0; p < 4; ++p) {
      int idx = w * 4 + p;
      int smt = idx & 1;
      int snt = idx >> 1;           // 0..15
      v8f sa = (v8f){};
#pragma unroll
      for (int kk = 0; kk < 2; ++kk) {
        int kbq = kk * 32;
        v16h af, bf;
        int qrow = smt * 16 + lr;
#pragma unroll
        for (int j = 0; j < 8; ++j) {
          af[j]     = Qs[qrow][kbq + ho + j];
          af[8 + j] = Qs[qrow][kbq + 16 + ho + j];
        }
        int key = k0 + snt * 16 + lr;
        const _Float16* krow =
            kp + (size_t)(b * Lk + key) * ks + hoff + kbq + ho;
        v8h lo = *(const v8h*)(krow);
        v8h hi = *(const v8h*)(krow + 16);
#pragma unroll
        for (int j = 0; j < 8; ++j) { bf[j] = lo[j]; bf[8 + j] = hi[j]; }
        sa = wmma_f16(af, bf, sa);
      }
      int scol = snt * 16 + lr;
#pragma unroll
      for (int r = 0; r < 8; ++r)
        Ss[smt * 16 + r + ho][scol] = sa[r];
    }
    __syncthreads();

    // ---- online softmax over the chunk -----------------------------------
    float lm = -1e30f;
    for (int cc = sub; cc < CH; cc += 8) {
      float s = Ss[row][cc] * scale;
      int gc = k0 + cc;
      bool ok;
      if (maskType == 0)      ok = true;
      else if (maskType == 1) ok = (gc <= qbase + row);
      else ok = (mbits[(size_t)(b * Lq + qbase + row) * mwrds + (gc >> 5)]
                 >> (gc & 31)) & 1u;
      if (!ok) s = -1e30f;
      Ss[row][cc] = s;
      lm = fmaxf(lm, s);
    }
    red[row][sub] = lm;
    __syncthreads();
    if (sub == 0) {
      float m = rowM[row];
#pragma unroll
      for (int t = 0; t < 8; ++t) m = fmaxf(m, red[row][t]);
      rowSc[row] = __expf(rowM[row] - m);
      rowM[row]  = m;
    }
    __syncthreads();

    float es = 0.f;
    float rm = rowM[row];
    for (int cc = sub; cc < CH; cc += 8) {
      float e = __expf(Ss[row][cc] - rm);
      Ps[row][cc] = (_Float16)e;
      es += e;
    }
    red[row][sub] = es;
    __syncthreads();
    if (sub == 0) {
      float s2 = 0.f;
#pragma unroll
      for (int t = 0; t < 8; ++t) s2 += red[row][t];
      rowL[row] = rowL[row] * rowSc[row] + s2;
    }
    __syncthreads();

    // ---- rescale O accumulator, then O += P_chunk * V --------------------
#pragma unroll
    for (int r = 0; r < 8; ++r) co[r] *= rowSc[mt * 16 + r + ho];

#pragma unroll
    for (int kkk = 0; kkk < 8; ++kkk) {
      int kb2 = kkk * 32;
      v16h af, bf;
      int prow = mt * 16 + lr;
#pragma unroll
      for (int j = 0; j < 8; ++j) {
        af[j]     = Ps[prow][kb2 + ho + j];
        af[8 + j] = Ps[prow][kb2 + 16 + ho + j];
      }
      int dcol = nt * 16 + lr;
#pragma unroll
      for (int j = 0; j < 8; ++j) {
        int key0 = k0 + kb2 + ho + j;
        int key1 = k0 + kb2 + 16 + ho + j;
        bf[j]     = vp[(size_t)(b * Lk + key0) * vs + hoff + dcol];
        bf[8 + j] = vp[(size_t)(b * Lk + key1) * vs + hoff + dcol];
      }
      co = wmma_f16(af, bf, co);
    }
    __syncthreads();
  }

  // ---- finalize: divide by row sums, store 16x16 tile (f16) --------------
#pragma unroll
  for (int r = 0; r < 8; ++r) {
    int ml = mt * 16 + r + ho;                 // local row (0..31)
    float inv = 1.0f / rowL[ml];
    op[(size_t)(b * Lq + qbase + ml) * os + hoff + nt * 16 + lr] =
        (_Float16)(co[r] * inv);
  }
}

// ===========================================================================
// Host orchestration
// ===========================================================================
extern "C" void kernel_launch(void* const* d_in, const int* in_sizes, int n_in,
                              void* d_out, int out_size, void* d_ws, size_t ws_size,
                              hipStream_t stream) {
  (void)in_sizes; (void)n_in; (void)out_size; (void)ws_size;

  const float* enc_in   = (const float*)d_in[0];
  const float* dec_in   = (const float*)d_in[1];
  const float* enc_xyz  = (const float*)d_in[2];
  const float* e_qkv_w  = (const float*)d_in[3];
  const float* e_qkv_b  = (const float*)d_in[4];
  const float* e_out_w  = (const float*)d_in[5];
  const float* e_out_b  = (const float*)d_in[6];
  const float* e_ln1_w  = (const float*)d_in[7];
  const float* e_ln1_b  = (const float*)d_in[8];
  const float* e_ln2_w  = (const float*)d_in[9];
  const float* e_ln2_b  = (const float*)d_in[10];
  const float* e_ff1_w  = (const float*)d_in[11];
  const float* e_ff1_b  = (const float*)d_in[12];
  const float* e_ff2_w  = (const float*)d_in[13];
  const float* e_ff2_b  = (const float*)d_in[14];
  const float* d_sa_qkv_w = (const float*)d_in[15];
  const float* d_sa_qkv_b = (const float*)d_in[16];
  const float* d_sa_out_w = (const float*)d_in[17];
  const float* d_sa_out_b = (const float*)d_in[18];
  const float* d_ca_qkv_w = (const float*)d_in[19];
  const float* d_ca_qkv_b = (const float*)d_in[20];
  const float* d_ca_out_w = (const float*)d_in[21];
  const float* d_ca_out_b = (const float*)d_in[22];
  const float* d_ln1_w  = (const float*)d_in[23];
  const float* d_ln1_b  = (const float*)d_in[24];
  const float* d_ln2_w  = (const float*)d_in[25];
  const float* d_ln2_b  = (const float*)d_in[26];
  const float* d_ln3_w  = (const float*)d_in[27];
  const float* d_ln3_b  = (const float*)d_in[28];
  const float* d_ff1_w  = (const float*)d_in[29];
  const float* d_ff1_b  = (const float*)d_in[30];
  const float* d_ff2_w  = (const float*)d_in[31];
  const float* d_ff2_b  = (const float*)d_in[32];
  const float* enc_ln_w = (const float*)d_in[33];
  const float* enc_ln_b = (const float*)d_in[34];
  const float* dec_ln_w = (const float*)d_in[35];
  const float* dec_ln_b = (const float*)d_in[36];

  // Workspace layout (units of float; all f16 blocks have even counts and
  // start 16B-aligned for b128 async copies)
  float* ws = (float*)d_ws;
  size_t off = 0;
  unsigned* maskBits = (unsigned*)(ws + off); off += (size_t)MTOK * (LE_ / 32);
  float*    xbuf  = ws + off;              off += (size_t)MTOK * D_;
  _Float16* hbuf  = (_Float16*)(ws + off); off += (size_t)MTOK * D_ / 2;
  _Float16* qkv   = (_Float16*)(ws + off); off += (size_t)MTOK * 3 * D_ / 2;
  _Float16* attnb = (_Float16*)(ws + off); off += (size_t)MTOK * D_ / 2;
  _Float16* ffb   = (_Float16*)(ws + off); off += (size_t)MTOK * F_ / 2;
  _Float16* qcr   = (_Float16*)(ws + off); off += (size_t)MTOK * D_ / 2;
  _Float16* kvcr  = (_Float16*)(ws + off); off += (size_t)MTOK * 2 * D_ / 2;
  _Float16* encH  = (_Float16*)(ws + off); off += (size_t)MTOK * D_ / 2;
  _Float16* wptr  = (_Float16*)(ws + off);

  float* enc_out = (float*)d_out;                       // (B, LE, D)
  float* dec_out = (float*)d_out + (size_t)MTOK * D_;   // (B, LD, D)

  // --- convert all weights to f16 once ------------------------------------
  auto cvt = [&](const float* src, size_t n) {
    _Float16* dst = wptr; wptr += n;
    cvt_kernel<<<(unsigned)((n / 4 + 255) / 256), 256, 0, stream>>>(src, dst, (int)n);
    return dst;
  };
  _Float16* wEqkv = cvt(e_qkv_w,   (size_t)NE_ * 3 * D_ * D_);
  _Float16* wEout = cvt(e_out_w,   (size_t)NE_ * D_ * D_);
  _Float16* wEff1 = cvt(e_ff1_w,   (size_t)NE_ * F_ * D_);
  _Float16* wEff2 = cvt(e_ff2_w,   (size_t)NE_ * D_ * F_);
  _Float16* wDsaq = cvt(d_sa_qkv_w,(size_t)ND_ * 3 * D_ * D_);
  _Float16* wDsao = cvt(d_sa_out_w,(size_t)ND_ * D_ * D_);
  _Float16* wDcaq = cvt(d_ca_qkv_w,(size_t)ND_ * 3 * D_ * D_);
  _Float16* wDcao = cvt(d_ca_out_w,(size_t)ND_ * D_ * D_);
  _Float16* wDff1 = cvt(d_ff1_w,   (size_t)ND_ * F_ * D_);
  _Float16* wDff2 = cvt(d_ff2_w,   (size_t)ND_ * D_ * F_);

  auto gemm = [&](const _Float16* A, int lda, const _Float16* W,
                  const float* bias, const float* res,
                  float* Cf, _Float16* Ch, int ldc, int M, int N, int Kd, int act) {
    dim3 g((unsigned)(N / GBN), (unsigned)(M / GBM));
    gemm_kernel<<<g, 256, 0, stream>>>(A, lda, W, Kd, bias, res, Cf, Ch, ldc, act);
  };
  auto ln = [&](const float* x, const float* w, const float* b,
                float* of, _Float16* oh) {
    ln_kernel<<<MTOK / 8, 256, 0, stream>>>(x, w, b, of, oh, MTOK);
  };

  const float scale = 0.125f;   // 1/sqrt(64)
  dim3 agrid(LE_ / QT, H_, B_);

  // --- kNN allow-mask -----------------------------------------------------
  knn_kernel<<<(B_ * LE_ + 127) / 128, 128, 0, stream>>>(enc_xyz, maskBits);

  // --- Encoder ------------------------------------------------------------
  const float* xc = enc_in;
  for (int i = 0; i < NE_; ++i) {
    ln(xc, e_ln1_w + i * D_, e_ln1_b + i * D_, nullptr, hbuf);
    gemm(hbuf, D_, wEqkv + (size_t)i * 3 * D_ * D_, e_qkv_b + i * 3 * D_,
         nullptr, nullptr, qkv, 3 * D_, MTOK, 3 * D_, D_, 0);
    attn_kernel<<<agrid, 256, 0, stream>>>(qkv, 3 * D_, qkv + D_, 3 * D_,
                                           qkv + 2 * D_, 3 * D_, attnb, D_,
                                           LE_, LE_, 2, maskBits, scale);
    gemm(attnb, D_, wEout + (size_t)i * D_ * D_, e_out_b + i * D_,
         xc, xbuf, nullptr, D_, MTOK, D_, D_, 0);
    xc = xbuf;
    ln(xc, e_ln2_w + i * D_, e_ln2_b + i * D_, nullptr, hbuf);
    gemm(hbuf, D_, wEff1 + (size_t)i * F_ * D_, e_ff1_b + i * F_,
         nullptr, nullptr, ffb, F_, MTOK, F_, D_, 1);
    gemm(ffb, F_, wEff2 + (size_t)i * D_ * F_, e_ff2_b + i * D_,
         xbuf, xbuf, nullptr, D_, MTOK, D_, F_, 0);
  }
  ln(xbuf, enc_ln_w, enc_ln_b, enc_out, encH);   // f32 to d_out + f16 for cross-KV

  // --- Decoder ------------------------------------------------------------
  const float* yc = dec_in;
  for (int i = 0; i < ND_; ++i) {
    // self-attention (causal)
    ln(yc, d_ln1_w + i * D_, d_ln1_b + i * D_, nullptr, hbuf);
    gemm(hbuf, D_, wDsaq + (size_t)i * 3 * D_ * D_, d_sa_qkv_b + i * 3 * D_,
         nullptr, nullptr, qkv, 3 * D_, MTOK, 3 * D_, D_, 0);
    attn_kernel<<<agrid, 256, 0, stream>>>(qkv, 3 * D_, qkv + D_, 3 * D_,
                                           qkv + 2 * D_, 3 * D_, attnb, D_,
                                           LD_, LD_, 1, nullptr, scale);
    gemm(attnb, D_, wDsao + (size_t)i * D_ * D_, d_sa_out_b + i * D_,
         yc, xbuf, nullptr, D_, MTOK, D_, D_, 0);
    yc = xbuf;
    // cross-attention (no mask); K/V from enc_out (f16 copy)
    ln(yc, d_ln2_w + i * D_, d_ln2_b + i * D_, nullptr, hbuf);
    gemm(hbuf, D_, wDcaq + (size_t)i * 3 * D_ * D_, d_ca_qkv_b + i * 3 * D_,
         nullptr, nullptr, qcr, D_, MTOK, D_, D_, 0);                 // Q rows [0,D)
    gemm(encH, D_, wDcaq + (size_t)i * 3 * D_ * D_ + (size_t)D_ * D_,
         d_ca_qkv_b + i * 3 * D_ + D_,
         nullptr, nullptr, kvcr, 2 * D_, MTOK, 2 * D_, D_, 0);        // K,V rows [D,3D)
    attn_kernel<<<agrid, 256, 0, stream>>>(qcr, D_, kvcr, 2 * D_,
                                           kvcr + D_, 2 * D_, attnb, D_,
                                           LD_, LE_, 0, nullptr, scale);
    gemm(attnb, D_, wDcao + (size_t)i * D_ * D_, d_ca_out_b + i * D_,
         xbuf, xbuf, nullptr, D_, MTOK, D_, D_, 0);
    // feed-forward
    ln(xbuf, d_ln3_w + i * D_, d_ln3_b + i * D_, nullptr, hbuf);
    gemm(hbuf, D_, wDff1 + (size_t)i * F_ * D_, d_ff1_b + i * F_,
         nullptr, nullptr, ffb, F_, MTOK, F_, D_, 1);
    gemm(ffb, F_, wDff2 + (size_t)i * D_ * F_, d_ff2_b + i * D_,
         xbuf, xbuf, nullptr, D_, MTOK, D_, F_, 0);
  }
  ln(xbuf, dec_ln_w, dec_ln_b, dec_out, nullptr);
}
